// MinLSTMCell_13984413516109
// MI455X (gfx1250) — matrix operand order repaired and checked by reading, not compiled, not run on hardware
//
#include <hip/hip_runtime.h>

// ---------------------------------------------------------------------------
// minLSTM on MI455X (gfx1250, wave32).
//   Phase 0: convert x and the 3 weight matrices fp32 -> bf16 (RNE).
//   Phase 1: fused WMMA GEMM (V_WMMA_F32_16X16X32_BF16) computing all three
//            projections + sigmoid/normalize epilogue, storing packed
//            (a_t, b_t) float2 scan coefficients.
//            Per wave: 64(M) x 16(N) x 3 gemms -> 12 WMMAs per K-step,
//            each B tile reused by 4 WMMAs (A-reuse hides B-load latency).
//   Phase 2: sequential scan h_t = a_t*h_{t-1} + b_t, one thread per (b,h)
//            channel (6144 chains), unrolled x4.
// Workspace: x_bf16 (48MB) | 3 weights bf16 (3.4MB) | gates float2 (192MB).
// x in bf16 fits in the 192MB L2, so the 48 N-tile passes re-hit L2.
// ---------------------------------------------------------------------------

#define D_IN  768
#define H_OUT 768
#define BATCH 8
#define TSEQ  4096
#define MTOT  (BATCH * TSEQ)   // 32768

typedef __attribute__((ext_vector_type(16))) __bf16         v16bf;
typedef __attribute__((ext_vector_type(8)))  float          v8f;
typedef __attribute__((ext_vector_type(4)))  unsigned int   u32x4;
typedef __attribute__((ext_vector_type(8)))  unsigned short u16x8;

union AB16 { v16bf v; u32x4 q[2]; };

__device__ __forceinline__ unsigned short f32_to_bf16_rne(float f) {
  unsigned int u = __float_as_uint(f);
  u += 0x7FFFu + ((u >> 16) & 1u);      // round-to-nearest-even
  return (unsigned short)(u >> 16);
}

// --------------------------- Phase 0: fp32 -> bf16 --------------------------
__global__ __launch_bounds__(256)
void cvt_f32_to_bf16_kernel(const float* __restrict__ src,
                            unsigned short* __restrict__ dst, int n8) {
  int i = blockIdx.x * blockDim.x + threadIdx.x;
  if (i >= n8) return;
  const float* s = src + (size_t)i * 8;
  u16x8 r;
#pragma unroll
  for (int j = 0; j < 8; ++j) r[j] = f32_to_bf16_rne(s[j]);
  *reinterpret_cast<u16x8*>(dst + (size_t)i * 8) = r;  // 16B store
}

// ----------------- Phase 1: fused 3-way WMMA GEMM + gate math ---------------
__device__ __forceinline__ float2 gate_combine(float fz, float iz, float hz) {
  float fs = 1.0f / (1.0f + __expf(-fz));
  float is = 1.0f / (1.0f + __expf(-iz));
  float den = fs + is + 1e-8f;
  float2 r;
  r.x = fs / den;            // a_t
  r.y = (is / den) * hz;     // b_t
  return r;
}

#define MSUB 4   // 16-row subtiles per wave => 64 rows of M per wave

__global__ __launch_bounds__(256)
void minlstm_gates_wmma_kernel(const unsigned short* __restrict__ xbf,
                               const unsigned short* __restrict__ wf,
                               const unsigned short* __restrict__ wi,
                               const unsigned short* __restrict__ wh,
                               const float* __restrict__ fb,
                               const float* __restrict__ ib,
                               const float* __restrict__ hb,
                               float2* __restrict__ gates) {
  const int lane  = threadIdx.x & 31;
  const int wave  = threadIdx.x >> 5;               // 8 waves / block
  const int mwave = blockIdx.x * (256 * MSUB / 4) * 4 + wave * (16 * MSUB);
  const int nbase = blockIdx.y * 16;
  const int l15   = lane & 15;
  const int lhi   = lane >> 4;                      // 0 or 1

  // A operand (16x32 bf16): lane = row m, two contiguous 16B K-runs
  //   vgpr0-3: K = lhi*8 .. +7 ; vgpr4-7: K = 16 + lhi*8 .. +7
  const unsigned short* arow[MSUB];
#pragma unroll
  for (int s = 0; s < MSUB; ++s)
    arow[s] = xbf + (size_t)(mwave + s * 16 + l15) * D_IN + lhi * 8;

  // B operand (32x16 bf16): lane = col n -> two contiguous 16B K-runs of W row n
  //   vgpr0-3: K = lhi*8 .. +7 ; vgpr4-7: K = 16 + lhi*8 .. +7
  const size_t boff = (size_t)(nbase + l15) * D_IN + (size_t)(lhi * 8);
  const unsigned short* brf = wf + boff;
  const unsigned short* bri = wi + boff;
  const unsigned short* brh = wh + boff;

  v8f accF[MSUB], accI[MSUB], accH[MSUB];
#pragma unroll
  for (int s = 0; s < MSUB; ++s) { accF[s] = {}; accI[s] = {}; accH[s] = {}; }

  for (int k0 = 0; k0 < D_IN; k0 += 32) {
    AB16 a[MSUB], bF, bI, bH;
#pragma unroll
    for (int s = 0; s < MSUB; ++s) {
      a[s].q[0] = *reinterpret_cast<const u32x4*>(arow[s] + k0);
      a[s].q[1] = *reinterpret_cast<const u32x4*>(arow[s] + k0 + 16);
    }
    bF.q[0] = *reinterpret_cast<const u32x4*>(brf + k0);
    bF.q[1] = *reinterpret_cast<const u32x4*>(brf + k0 + 16);
    bI.q[0] = *reinterpret_cast<const u32x4*>(bri + k0);
    bI.q[1] = *reinterpret_cast<const u32x4*>(bri + k0 + 16);
    bH.q[0] = *reinterpret_cast<const u32x4*>(brh + k0);
    bH.q[1] = *reinterpret_cast<const u32x4*>(brh + k0 + 16);

#pragma unroll
    for (int s = 0; s < MSUB; ++s)
      accF[s] = __builtin_amdgcn_wmma_f32_16x16x32_bf16(
          false, a[s].v, false, bF.v, (short)0, accF[s], false, false);
#pragma unroll
    for (int s = 0; s < MSUB; ++s)
      accI[s] = __builtin_amdgcn_wmma_f32_16x16x32_bf16(
          false, a[s].v, false, bI.v, (short)0, accI[s], false, false);
#pragma unroll
    for (int s = 0; s < MSUB; ++s)
      accH[s] = __builtin_amdgcn_wmma_f32_16x16x32_bf16(
          false, a[s].v, false, bH.v, (short)0, accH[s], false, false);
  }

  // C/D layout: m = vgpr + 8*lhi (+16 per subtile), n = lane%16
  const int   ncol  = nbase + l15;
  const float fbias = fb[ncol];
  const float ibias = ib[ncol];
  const float hbias = hb[ncol];
  const int   moff  = mwave + lhi * 8;

#pragma unroll
  for (int s = 0; s < MSUB; ++s) {
#pragma unroll
    for (int v = 0; v < 8; ++v) {
      size_t m = (size_t)(moff + s * 16 + v);
      gates[m * H_OUT + ncol] = gate_combine(accF[s][v] + fbias,
                                             accI[s][v] + ibias,
                                             accH[s][v] + hbias);
    }
  }
}

// ------------------------ Phase 2: linear recurrence ------------------------
__global__ __launch_bounds__(256)
void minlstm_scan_kernel(const float2* __restrict__ gates,
                         const float* __restrict__ h0,
                         float* __restrict__ out) {
  int c = blockIdx.x * blockDim.x + threadIdx.x;   // channel = b*H + h
  if (c >= BATCH * H_OUT) return;
  const int b  = c / H_OUT;
  const int hh = c - b * H_OUT;

  float h = h0[c];                                  // h0 is (B,1,H) flat
  const float2* g = gates + (size_t)b * TSEQ * H_OUT + hh;
  float*        o = out   + (size_t)b * TSEQ * H_OUT + hh;

  for (int t = 0; t < TSEQ; t += 4) {
    float2 g0 = g[(size_t)(t + 0) * H_OUT];
    float2 g1 = g[(size_t)(t + 1) * H_OUT];
    float2 g2 = g[(size_t)(t + 2) * H_OUT];
    float2 g3 = g[(size_t)(t + 3) * H_OUT];
    h = fmaf(g0.x, h, g0.y); o[(size_t)(t + 0) * H_OUT] = h;
    h = fmaf(g1.x, h, g1.y); o[(size_t)(t + 1) * H_OUT] = h;
    h = fmaf(g2.x, h, g2.y); o[(size_t)(t + 2) * H_OUT] = h;
    h = fmaf(g3.x, h, g3.y); o[(size_t)(t + 3) * H_OUT] = h;
  }
}

// ------------------------------- launcher -----------------------------------
extern "C" void kernel_launch(void* const* d_in, const int* in_sizes, int n_in,
                              void* d_out, int out_size, void* d_ws, size_t ws_size,
                              hipStream_t stream) {
  const float* x   = (const float*)d_in[0];
  const float* h0  = (const float*)d_in[1];
  const float* f_w = (const float*)d_in[2];
  const float* f_b = (const float*)d_in[3];
  const float* i_w = (const float*)d_in[4];
  const float* i_b = (const float*)d_in[5];
  const float* h_w = (const float*)d_in[6];
  const float* h_b = (const float*)d_in[7];
  float* out = (float*)d_out;

  char* ws = (char*)d_ws;
  const size_t xbf_elems = (size_t)MTOT * D_IN;      // 25165824
  const size_t w_elems   = (size_t)H_OUT * D_IN;     // 589824
  unsigned short* xbf = (unsigned short*)ws;
  unsigned short* wfb = xbf + xbf_elems;
  unsigned short* wib = wfb + w_elems;
  unsigned short* whb = wib + w_elems;
  float2* gates = (float2*)(ws + 2 * (xbf_elems + 3 * w_elems)); // 53870592 B, 256-aligned

  // Phase 0: bf16 conversions (8 elems/thread, 16B stores)
  {
    int n8 = (int)(xbf_elems / 8);
    cvt_f32_to_bf16_kernel<<<(n8 + 255) / 256, 256, 0, stream>>>(x, xbf, n8);
    int w8 = (int)(w_elems / 8);
    cvt_f32_to_bf16_kernel<<<(w8 + 255) / 256, 256, 0, stream>>>(f_w, wfb, w8);
    cvt_f32_to_bf16_kernel<<<(w8 + 255) / 256, 256, 0, stream>>>(i_w, wib, w8);
    cvt_f32_to_bf16_kernel<<<(w8 + 255) / 256, 256, 0, stream>>>(h_w, whb, w8);
  }

  // Phase 1: fused WMMA GEMM + gate epilogue
  // 8 waves/block * 64 rows/wave = 512 rows of M per block
  {
    dim3 grid(MTOT / (16 * MSUB * 8), H_OUT / 16, 1);   // 64 x 48
    minlstm_gates_wmma_kernel<<<grid, 256, 0, stream>>>(
        xbf, wfb, wib, whb, f_b, i_b, h_b, gates);
  }

  // Phase 2: scan over T
  {
    int nchan = BATCH * H_OUT;              // 6144
    minlstm_scan_kernel<<<(nchan + 255) / 256, 256, 0, stream>>>(gates, h0, out);
  }
}
